// MFA_12455405159156
// MI455X (gfx1250) — matrix operand order repaired and checked by reading, not compiled, not run on hardware
//
#include <hip/hip_runtime.h>
#include <math.h>

// ---------------------------------------------------------------------------
// MFA one-EM-step for gfx1250 (MI455X). bf16 WMMA (v_wmma_f32_16x16x32_bf16)
// for all large contractions, f32 everywhere else.
// N=16384, d=1024, K=32, l=16 (hardcoded to match the reference setup).
// GEMM waves now compute 1x{2,4} tile strips: the A fragment is loaded once
// and reused across NT WMMAs, quadrupling FLOP/byte out of L2.
// ---------------------------------------------------------------------------

#define NN   16384
#define DD   1024
#define KC   32
#define LL   16
#define J1   544            // 512 (K*l columns of B) + 32 (cross columns)
#define LOG2PI_F 1.8378770664093453f

typedef __attribute__((ext_vector_type(16))) __bf16 v16bf;
typedef __attribute__((ext_vector_type(8)))  float  v8f;
typedef __attribute__((ext_vector_type(4)))  unsigned int u32x4;

union FragBF { v16bf v; u32x4 q[2]; };

__device__ inline unsigned short f32_to_bf16(float f) {
  unsigned int u = __builtin_bit_cast(unsigned int, f);
  unsigned int r = u + 0x7FFFu + ((u >> 16) & 1u);
  return (unsigned short)(r >> 16);
}

// ---------------------------------------------------------------------------
// Generic bf16 GEMM: C[M,Ncols] = A[M,Kd] * B[Kd,Ncols], f32 accumulate.
// One wave per 16 x (16*NT) output strip: A fragment loaded once per K-step,
// reused for NT B tiles. grid = (Ncols/(16*NT), M/16), block = 32.
// ---------------------------------------------------------------------------
template <int NT>
__global__ void gemm_bf16_kernel(const unsigned short* __restrict__ A,
                                 const unsigned short* __restrict__ B,
                                 float* __restrict__ C,
                                 int M, int Ncols, int Kd) {
  const int ntbase = blockIdx.x * NT;      // first N tile of this wave's strip
  const int mtile  = blockIdx.y;
  const int lane   = threadIdx.x & 31;
  const int mrow   = lane & 15;
  const int kh     = lane >> 4;            // which K-half this lane holds for A
  const unsigned short* Arow = A + (size_t)(mtile * 16 + mrow) * Kd;

  v8f acc[NT];
#pragma unroll
  for (int i = 0; i < NT; ++i)
    acc[i] = (v8f){0.f, 0.f, 0.f, 0.f, 0.f, 0.f, 0.f, 0.f};

  for (int kk = 0; kk < Kd; kk += 32) {
    FragBF fa;
    // A 16x32 bf16 fragment: lane holds M=mrow; elems 0..7 = K kh*8+0..7,
    // elems 8..15 = K 16+kh*8+0..7 (two contiguous 16B runs).
    fa.q[0] = *(const u32x4*)(Arow + kk + kh * 8);
    fa.q[1] = *(const u32x4*)(Arow + kk + 16 + kh * 8);
    // B 32x16 bf16 fragments: lane holds column N = lane&15; elems 0..7 = K kh*8+0..7,
    // elems 8..15 = K 16+kh*8+0..7 (same K order as A).
    const unsigned short* brow = B + (size_t)kk * Ncols + ntbase * 16 + (lane & 15);
#pragma unroll
    for (int i = 0; i < NT; ++i) {
      FragBF fb;
#pragma unroll
      for (int s = 0; s < 16; ++s) {
        const int kr = (s < 8) ? (kh * 8 + s) : (16 + kh * 8 + (s - 8));
        fb.v[s] = __builtin_bit_cast(__bf16, brow[(size_t)kr * Ncols + i * 16]);
      }
      acc[i] = __builtin_amdgcn_wmma_f32_16x16x32_bf16(false, fa.v, false, fb.v,
                                                       (short)0, acc[i], false, false);
    }
  }
  // C/D layout: VGPR r -> M = r + 8*(lane>>4); N = lane&15
  const int mbase = mtile * 16 + (lane >> 4) * 8;
#pragma unroll
  for (int i = 0; i < NT; ++i) {
    const int ncol = (ntbase + i) * 16 + (lane & 15);
#pragma unroll
    for (int r = 0; r < 8; ++r)
      C[(size_t)(mbase + r) * Ncols + ncol] = acc[i][r];
  }
}

// ---------------------------------------------------------------------------
// Shared-memory Gauss-Jordan inverse of a 16x16 matrix (augmented, 16x33
// layout, cols 0..15 = M, cols 16..31 = identity -> inverse). 256 threads.
// ---------------------------------------------------------------------------
__device__ void invert16(float* sh, int t, float* s_logdet) {
  const int r = t & 15;
  const int c = t >> 4;          // 0..15 ; this thread owns cols c and c+16
  for (int piv = 0; piv < 16; ++piv) {
    __syncthreads();
    float pv = sh[piv * 33 + piv];
    if (s_logdet && t == 0) *s_logdet += logf(fabsf(pv));
    float invp = 1.0f / pv;
    __syncthreads();
    if (t < 32) sh[piv * 33 + t] *= invp;
    __syncthreads();
    float factor = (r == piv) ? 0.0f : sh[r * 33 + piv];
    float p0 = sh[piv * 33 + c];
    float p1 = sh[piv * 33 + c + 16];
    __syncthreads();
    sh[r * 33 + c]      -= factor * p0;
    sh[r * 33 + c + 16] -= factor * p1;
  }
  __syncthreads();
}

// ---------------------------------------------------------------------------
// K0: zero accumulators + softmax(PI_logits). 1 block x 512 threads.
// ---------------------------------------------------------------------------
__global__ void init_kernel(const float* __restrict__ logits, float* PI,
                            float* sumRQ, float* r_sum, float* trxsq, float* t1v) {
  int t = threadIdx.x;
  sumRQ[t] = 0.f;                      // 512 entries
  if (t < KC) {
    r_sum[t] = 0.f; trxsq[t] = 0.f; t1v[t] = 0.f;
    float v = logits[t];
    float mx = v;
    for (int off = 16; off > 0; off >>= 1) mx = fmaxf(mx, __shfl_xor(mx, off, 32));
    float e = expf(v - mx);
    float s = e;
    for (int off = 16; off > 0; off >>= 1) s += __shfl_xor(s, off, 32);
    PI[t] = e / s;
  }
}

// ---------------------------------------------------------------------------
// K1: pack x -> bf16 (row major), x^2 -> bf16, x^T -> bf16.
// ---------------------------------------------------------------------------
__global__ void convert_kernel(const float* __restrict__ x,
                               unsigned short* xbf, unsigned short* x2bf,
                               unsigned short* xtbf) {
  size_t idx = (size_t)blockIdx.x * 256 + threadIdx.x;   // < N*D
  float xv = x[idx];
  xbf[idx]  = f32_to_bf16(xv);
  x2bf[idx] = f32_to_bf16(xv * xv);
  size_t n = idx >> 10, d = idx & 1023;
  xtbf[d * (size_t)NN + n] = f32_to_bf16(xv);
}

// K1b: x_sq[n] = sum_d x[n,d]^2 (f32). One block per row.
__global__ void xsq_kernel(const float* __restrict__ x, float* x_sq) {
  __shared__ float red[256];
  int n = blockIdx.x, t = threadIdx.x;
  const float* xn = x + (size_t)n * DD;
  float s = 0.f;
  for (int d = t; d < DD; d += 256) { float v = xn[d]; s += v * v; }
  red[t] = s; __syncthreads();
  for (int off = 128; off > 0; off >>= 1) {
    if (t < off) red[t] += red[t + off];
    __syncthreads();
  }
  if (t == 0) x_sq[n] = red[0];
}

// ---------------------------------------------------------------------------
// K2 prep (one block per component k):
//  - pack weight matrices: Wbig[d, k*16+l] = (iD*A), Wbig[d, 512+k] = iD*MU,
//    W2[d,k] = iD, Abfw[d, k*16+l] = A    (all bf16, column blocks disjoint)
//  - bmu[k,l] = sum_d MU*iD*A ; constk = sum_d iD*MU^2 ; sum(log_D)
//  - L = I + A^T (iD*A) -> iL = inv(L), logdet(L)
//  - inv_M = inv(A^T A + s2 I),  s2 = exp(log_D[k,0])
// ---------------------------------------------------------------------------
__global__ void prep_kernel(const float* __restrict__ MU, const float* __restrict__ A,
                            const float* __restrict__ log_D,
                            unsigned short* Wbig, unsigned short* W2,
                            unsigned short* Abfw,
                            float* iL, float* invM, float* bmu,
                            float* constk, float* logdetS) {
  const int k = blockIdx.x, t = threadIdx.x;
  __shared__ float s_bmu[16];
  __shared__ float s_const, s_slogD, s_logdet;
  __shared__ float sh[16 * 33];
  if (t < 16) s_bmu[t] = 0.f;
  if (t == 0) { s_const = 0.f; s_slogD = 0.f; s_logdet = 0.f; }
  __syncthreads();

  const float* Ak  = A + (size_t)k * DD * LL;
  const float* MUk = MU + (size_t)k * DD;
  const float* lDk = log_D + (size_t)k * DD;

  float bmu_loc[16];
#pragma unroll
  for (int l = 0; l < 16; ++l) bmu_loc[l] = 0.f;
  float const_loc = 0.f, slog_loc = 0.f;

  for (int d = t; d < DD; d += 256) {
    float ld = lDk[d];
    float iD = expf(-ld);
    float mu = MUk[d];
#pragma unroll
    for (int l = 0; l < 16; ++l) {
      float a = Ak[d * 16 + l];
      float b = iD * a;
      Wbig[(size_t)d * J1 + k * 16 + l] = f32_to_bf16(b);
      Abfw[(size_t)d * 512 + k * 16 + l] = f32_to_bf16(a);
      bmu_loc[l] += mu * b;
    }
    Wbig[(size_t)d * J1 + 512 + k] = f32_to_bf16(iD * mu);
    W2[(size_t)d * 32 + k] = f32_to_bf16(iD);
    const_loc += iD * mu * mu;
    slog_loc  += ld;
  }
#pragma unroll
  for (int l = 0; l < 16; ++l) atomicAdd(&s_bmu[l], bmu_loc[l]);
  atomicAdd(&s_const, const_loc);
  atomicAdd(&s_slogD, slog_loc);

  // A^T (iD A) and A^T A ; thread (l,m)
  const int l = t >> 4, m = t & 15;
  float atb = 0.f, ata = 0.f;
  for (int d = 0; d < DD; ++d) {
    float iD = expf(-lDk[d]);
    float al = Ak[d * 16 + l];
    float am = Ak[d * 16 + m];
    atb += al * iD * am;
    ata += al * am;
  }
  sh[l * 33 + m]      = (l == m ? 1.f : 0.f) + atb;   // L
  sh[l * 33 + 16 + m] = (l == m ? 1.f : 0.f);
  __syncthreads();
  invert16(sh, t, &s_logdet);
  iL[k * 256 + l * 16 + m] = sh[l * 33 + 16 + m];
  __syncthreads();

  float s2 = expf(lDk[0]);
  sh[l * 33 + m]      = ata + (l == m ? s2 : 0.f);    // A^T A + s2 I
  sh[l * 33 + 16 + m] = (l == m ? 1.f : 0.f);
  __syncthreads();
  invert16(sh, t, nullptr);
  invM[k * 256 + l * 16 + m] = sh[l * 33 + 16 + m];

  if (t < 16) bmu[k * 16 + t] = s_bmu[t];
  if (t == 0) { constk[k] = s_const; logdetS[k] = s_logdet + s_slogD; }
}

// ---------------------------------------------------------------------------
// K4: E-step finalize. One wave per sample n, lane = component k.
//   term2 via iL quadratic form; softmax across the wave; accumulate r_sum,
//   r.T@x_sq ; emit r (f32) and r^T (bf16, [K,N]) for the mu GEMM.
// ---------------------------------------------------------------------------
__global__ void efin_kernel(const float* __restrict__ C1, const float* __restrict__ C2,
                            const float* __restrict__ bmu, const float* __restrict__ iL,
                            const float* __restrict__ constk, const float* __restrict__ logdetS,
                            const float* __restrict__ PI, const float* __restrict__ x_sq,
                            float* r_out, unsigned short* rTbf,
                            float* r_sum, float* trxsq) {
  __shared__ float s_rs[32], s_tx[32];
  int t = threadIdx.x;
  if (t < 32) { s_rs[t] = 0.f; s_tx[t] = 0.f; }
  __syncthreads();

  int wave = t >> 5, k = t & 31;
  int n = blockIdx.x * 8 + wave;
  const float* c1 = C1 + (size_t)n * J1;

  float y[16];
#pragma unroll
  for (int l = 0; l < 16; ++l) y[l] = c1[k * 16 + l] - bmu[k * 16 + l];
  float term2 = 0.f;
  const float* iLk = iL + k * 256;
#pragma unroll
  for (int l = 0; l < 16; ++l) {
    float s = 0.f;
#pragma unroll
    for (int m = 0; m < 16; ++m) s += iLk[l * 16 + m] * y[m];
    term2 += y[l] * s;
  }
  float term1 = C2[(size_t)n * 32 + k] - 2.f * c1[512 + k] + constk[k];
  float m_d = term1 - term2;
  float comp = PI[k] - 0.5f * ((float)DD * LOG2PI_F + logdetS[k] + m_d);

  float mx = comp;
  for (int off = 16; off > 0; off >>= 1) mx = fmaxf(mx, __shfl_xor(mx, off, 32));
  float e = expf(comp - mx);
  float ssum = e;
  for (int off = 16; off > 0; off >>= 1) ssum += __shfl_xor(ssum, off, 32);
  float r = e / ssum;

  r_out[(size_t)n * 32 + k] = r;
  rTbf[(size_t)k * NN + n] = f32_to_bf16(r);
  atomicAdd(&s_rs[k], r);
  atomicAdd(&s_tx[k], r * x_sq[n]);
  __syncthreads();
  if (t < 32) { atomicAdd(&r_sum[t], s_rs[t]); atomicAdd(&trxsq[t], s_tx[t]); }
}

// K6a: mu_new = Cmu / r_sum  (written straight into d_out).
__global__ void mudiv_kernel(const float* __restrict__ Cmu,
                             const float* __restrict__ r_sum, float* outMU) {
  int idx = blockIdx.x * 256 + threadIdx.x;          // < K*D
  int k = idx >> 10;
  outMU[idx] = Cmu[idx] / r_sum[k];
}

// K6b: qmu[k,l] = sum_d mu_new*A ; musq[k] = ||mu_new||^2. One block per k.
__global__ void qmu_kernel(const float* __restrict__ outMU, const float* __restrict__ A,
                           float* qmu, float* musq) {
  int k = blockIdx.x, t = threadIdx.x;
  __shared__ float s_q[16];
  __shared__ float s_m;
  if (t < 16) s_q[t] = 0.f;
  if (t == 0) s_m = 0.f;
  __syncthreads();
  int l = t & 15, grp = t >> 4;
  const float* muk = outMU + (size_t)k * DD;
  const float* Ak  = A + (size_t)k * DD * LL;
  float part = 0.f;
  for (int d = grp; d < DD; d += 16) part += muk[d] * Ak[d * 16 + l];
  atomicAdd(&s_q[l], part);
  float pm = 0.f;
  for (int d = t; d < DD; d += 256) { float m = muk[d]; pm += m * m; }
  atomicAdd(&s_m, pm);
  __syncthreads();
  if (t < 16) qmu[k * 16 + t] = s_q[t];
  if (t == 0) musq[k] = s_m;
}

// K8: RQ = r * (CQ - qmu) -> bf16 [N,512] ; sumRQ[j] accumulated.
__global__ void rq_kernel(const float* __restrict__ CQ, const float* __restrict__ qmu,
                          const float* __restrict__ r_out,
                          unsigned short* RQbf, float* sumRQ) {
  int t = threadIdx.x;
  int j = blockIdx.x * 256 + t;                      // column 0..511
  int n0 = blockIdx.y * 64;
  float qm = qmu[j];
  float acc = 0.f;
  for (int nn = 0; nn < 64; ++nn) {
    int n = n0 + nn;
    float q  = CQ[(size_t)n * 512 + j] - qm;
    float rq = r_out[(size_t)n * 32 + (j >> 4)] * q;
    RQbf[(size_t)n * 512 + j] = f32_to_bf16(rq);
    acc += rq;
  }
  atomicAdd(&sumRQ[j], acc);
}

// K10: SiAi[k,d,l] = (Sraw[d, k*16+l] - mu[k,d]*sumRQ[k,l]) / r_sum[k]
__global__ void siai_kernel(const float* __restrict__ Sraw, const float* __restrict__ outMU,
                            const float* __restrict__ sumRQ, const float* __restrict__ r_sum,
                            float* SiAi) {
  int idx = blockIdx.x * 256 + threadIdx.x;          // < K*D*16
  int k = idx >> 14, d = (idx >> 4) & 1023, l = idx & 15;
  SiAi[idx] = (Sraw[(size_t)d * 512 + k * 16 + l]
               - outMU[(size_t)k * DD + d] * sumRQ[k * 16 + l]) / r_sum[k];
}

// K11: per-k: ATSiAi = A^T SiAi ; H = invM*ATSiAi ; G = inv(s2 I + H).
__global__ void perk_kernel(const float* __restrict__ A, const float* __restrict__ log_D,
                            const float* __restrict__ SiAi, const float* __restrict__ invM,
                            float* G) {
  int k = blockIdx.x, t = threadIdx.x;
  int l = t >> 4, m = t & 15;
  __shared__ float ats[16 * 17];
  __shared__ float sh[16 * 33];
  const float* Ak = A + (size_t)k * DD * LL;
  const float* Sk = SiAi + (size_t)k * DD * LL;
  float acc = 0.f;
  for (int d = 0; d < DD; ++d)
    acc += Ak[d * 16 + l] * Sk[d * 16 + m];
  ats[l * 17 + m] = acc;
  __syncthreads();
  const float* iMk = invM + k * 256;
  float h = 0.f;
#pragma unroll
  for (int p = 0; p < 16; ++p) h += iMk[l * 16 + p] * ats[p * 17 + m];
  float s2 = expf(log_D[(size_t)k * DD]);
  sh[l * 33 + m]      = h + (l == m ? s2 : 0.f);
  sh[l * 33 + 16 + m] = (l == m ? 1.f : 0.f);
  __syncthreads();
  invert16(sh, t, nullptr);
  G[k * 256 + l * 16 + m] = sh[l * 33 + 16 + m];
}

// K12: A_new = SiAi @ G (into d_out) ; t1[k] += sum A_new * (SiAi @ invM).
__global__ void anew_kernel(const float* __restrict__ SiAi, const float* __restrict__ G,
                            const float* __restrict__ invM, float* outA, float* t1) {
  int k = blockIdx.y;
  int d = blockIdx.x * 256 + threadIdx.x;
  const float* S  = SiAi + ((size_t)k * DD + d) * 16;
  const float* Gk = G + k * 256;
  const float* Mk = invM + k * 256;
  float sv[16];
#pragma unroll
  for (int m = 0; m < 16; ++m) sv[m] = S[m];
  float dot = 0.f;
#pragma unroll
  for (int l = 0; l < 16; ++l) {
    float an = 0.f, si = 0.f;
#pragma unroll
    for (int m = 0; m < 16; ++m) { an += sv[m] * Gk[m * 16 + l]; si += sv[m] * Mk[m * 16 + l]; }
    outA[((size_t)k * DD + d) * 16 + l] = an;
    dot += an * si;
  }
  __shared__ float red[256];
  red[threadIdx.x] = dot;
  __syncthreads();
  for (int off = 128; off > 0; off >>= 1) {
    if (threadIdx.x < off) red[threadIdx.x] += red[threadIdx.x + off];
    __syncthreads();
  }
  if (threadIdx.x == 0) atomicAdd(&t1[k], red[0]);
}

// K13: log_D_new (broadcast per k) into d_out.
__global__ void logd_kernel(const float* __restrict__ trxsq, const float* __restrict__ r_sum,
                            const float* __restrict__ musq, const float* __restrict__ t1,
                            float* outLD) {
  int idx = blockIdx.x * 256 + threadIdx.x;          // < K*D
  int k = idx >> 10;
  float rs = r_sum[k];
  float ts = (trxsq[k] - rs * musq[k]) / rs;
  float sig = (ts - t1[k]) / (float)DD;
  outLD[idx] = logf(sig);
}

// K14: PI_logits_new = log(r_sum / sum(r_sum)). One wave.
__global__ void pik_kernel(const float* __restrict__ r_sum, float* outPI) {
  int k = threadIdx.x;
  float v = r_sum[k];
  float s = v;
  for (int off = 16; off > 0; off >>= 1) s += __shfl_xor(s, off, 32);
  outPI[k] = logf(v / s);
}

// ---------------------------------------------------------------------------
extern "C" void kernel_launch(void* const* d_in, const int* in_sizes, int n_in,
                              void* d_out, int out_size, void* d_ws, size_t ws_size,
                              hipStream_t stream) {
  (void)in_sizes; (void)n_in; (void)out_size; (void)ws_size;
  const float* x   = (const float*)d_in[0];
  const float* MU  = (const float*)d_in[1];
  const float* A   = (const float*)d_in[2];
  const float* lD  = (const float*)d_in[3];
  const float* PIl = (const float*)d_in[4];
  float* out = (float*)d_out;
  float* outMU = out;                                // [K,D]
  float* outA  = out + KC * DD;                      // [K,D,L]
  float* outLD = out + KC * DD + KC * DD * LL;       // [K,D]
  float* outPI = outLD + KC * DD;                    // [K]

  char* ws = (char*)d_ws;
  size_t cur = 0;
  auto alloc = [&](size_t bytes) { size_t o = cur; cur += (bytes + 255) & ~(size_t)255; return o; };

  size_t XBF   = alloc((size_t)NN * DD * 2);         // x bf16
  size_t XTBF  = alloc((size_t)NN * DD * 2);         // x^T bf16
  size_t XSQ   = alloc((size_t)NN * 4);              // ||x_n||^2
  size_t WBIG  = alloc((size_t)DD * J1 * 2);         // [iD*A | iD*MU] bf16
  size_t W2    = alloc((size_t)DD * 32 * 2);         // iD bf16
  size_t ABFW  = alloc((size_t)DD * 512 * 2);        // A bf16 (column blocked)
  size_t C1    = alloc((size_t)NN * J1 * 4);         // x@Wbig   (aliases CQ)
  size_t C2    = alloc((size_t)NN * 32 * 4);         // x^2 @ iD
  size_t RR    = alloc((size_t)NN * 32 * 4);         // responsibilities f32
  size_t RTBF  = alloc((size_t)KC * NN * 2);         // r^T bf16
  size_t CMU   = alloc((size_t)KC * DD * 4);         // r^T @ x
  size_t X2BF  = alloc((size_t)NN * DD * 2);         // x^2 bf16 (aliases RQbf)
  size_t SRAW  = alloc((size_t)DD * 512 * 4);        // x^T @ RQ
  size_t SIAI  = alloc((size_t)KC * DD * LL * 4);
  size_t IL    = alloc((size_t)KC * 256 * 4);
  size_t INVM  = alloc((size_t)KC * 256 * 4);
  size_t GMAT  = alloc((size_t)KC * 256 * 4);
  size_t BMU   = alloc((size_t)KC * 16 * 4);
  size_t CONST = alloc((size_t)KC * 4);
  size_t LDET  = alloc((size_t)KC * 4);
  size_t PIWS  = alloc((size_t)KC * 4);
  size_t QMU   = alloc((size_t)512 * 4);
  size_t SUMRQ = alloc((size_t)512 * 4);
  size_t RSUM  = alloc((size_t)KC * 4);
  size_t TRX   = alloc((size_t)KC * 4);
  size_t T1    = alloc((size_t)KC * 4);
  size_t MUSQ  = alloc((size_t)KC * 4);
  size_t CQ    = C1;                                 // lifetime-disjoint alias
  size_t RQBF  = X2BF;                               // lifetime-disjoint alias

  unsigned short* xbf  = (unsigned short*)(ws + XBF);
  unsigned short* xtbf = (unsigned short*)(ws + XTBF);
  unsigned short* x2bf = (unsigned short*)(ws + X2BF);
  unsigned short* wbig = (unsigned short*)(ws + WBIG);
  unsigned short* w2   = (unsigned short*)(ws + W2);
  unsigned short* abfw = (unsigned short*)(ws + ABFW);
  unsigned short* rtbf = (unsigned short*)(ws + RTBF);
  unsigned short* rqbf = (unsigned short*)(ws + RQBF);
  float* c1    = (float*)(ws + C1);
  float* c2    = (float*)(ws + C2);
  float* cq    = (float*)(ws + CQ);
  float* rr    = (float*)(ws + RR);
  float* cmu   = (float*)(ws + CMU);
  float* sraw  = (float*)(ws + SRAW);
  float* siai  = (float*)(ws + SIAI);
  float* il    = (float*)(ws + IL);
  float* invm  = (float*)(ws + INVM);
  float* gmat  = (float*)(ws + GMAT);
  float* bmu   = (float*)(ws + BMU);
  float* cns   = (float*)(ws + CONST);
  float* ldet  = (float*)(ws + LDET);
  float* piws  = (float*)(ws + PIWS);
  float* qmu   = (float*)(ws + QMU);
  float* srq   = (float*)(ws + SUMRQ);
  float* rsum  = (float*)(ws + RSUM);
  float* trx   = (float*)(ws + TRX);
  float* t1v   = (float*)(ws + T1);
  float* musq  = (float*)(ws + MUSQ);
  float* xsq   = (float*)(ws + XSQ);

  init_kernel<<<1, 512, 0, stream>>>(PIl, piws, srq, rsum, trx, t1v);
  convert_kernel<<<(NN * DD) / 256, 256, 0, stream>>>(x, xbf, x2bf, xtbf);
  xsq_kernel<<<NN, 256, 0, stream>>>(x, xsq);
  prep_kernel<<<KC, 256, 0, stream>>>(MU, A, lD, wbig, w2, abfw, il, invm, bmu, cns, ldet);

  // E-step GEMMs: C1 = x @ [iD*A | iD*MU] (Ncols=544, NT=2) ; C2 = x^2 @ iD
  gemm_bf16_kernel<2><<<dim3(J1 / 32, NN / 16), 32, 0, stream>>>(xbf, wbig, c1, NN, J1, DD);
  gemm_bf16_kernel<2><<<dim3(1, NN / 16), 32, 0, stream>>>(x2bf, w2, c2, NN, 32, DD);
  efin_kernel<<<NN / 8, 256, 0, stream>>>(c1, c2, bmu, il, cns, ldet, piws, xsq,
                                          rr, rtbf, rsum, trx);

  // mu_new = (r^T @ x) / r_sum   (Ncols=1024, NT=4)
  gemm_bf16_kernel<4><<<dim3(DD / 64, KC / 16), 32, 0, stream>>>(rtbf, xbf, cmu, KC, DD, NN);
  mudiv_kernel<<<(KC * DD) / 256, 256, 0, stream>>>(cmu, rsum, outMU);
  qmu_kernel<<<KC, 256, 0, stream>>>(outMU, A, qmu, musq);

  // Q = x @ A ; RQ = r*(Q - mu@A) ; SiAi_raw = x^T @ RQ   (Ncols=512, NT=4)
  gemm_bf16_kernel<4><<<dim3(512 / 64, NN / 16), 32, 0, stream>>>(xbf, abfw, cq, NN, 512, DD);
  rq_kernel<<<dim3(2, NN / 64), 256, 0, stream>>>(cq, qmu, rr, rqbf, srq);
  gemm_bf16_kernel<4><<<dim3(512 / 64, DD / 16), 32, 0, stream>>>(xtbf, rqbf, sraw, DD, 512, NN);

  siai_kernel<<<(KC * DD * LL) / 256, 256, 0, stream>>>(sraw, outMU, srq, rsum, siai);
  perk_kernel<<<KC, 256, 0, stream>>>(A, lD, siai, invm, gmat);
  anew_kernel<<<dim3(DD / 256, KC), 256, 0, stream>>>(siai, gmat, invm, outA, t1v);
  logd_kernel<<<(KC * DD) / 256, 256, 0, stream>>>(trx, rsum, musq, t1v, outLD);
  pik_kernel<<<1, 32, 0, stream>>>(rsum, outPI);
}
